// refinment_decoder_8246337208836
// MI455X (gfx1250) — matrix operand order repaired and checked by reading, not compiled, not run on hardware
//
#include <hip/hip_runtime.h>
#include <hip/hip_bf16.h>

// ---------------------------------------------------------------------------
// Pipeline (B=4, C=128, 64x64 -> 256x256, NC=2), f32 in / f32 out:
//   K-1 zeropad  : zero 1-px border of padded channels-last fusion buffer
//   K0 chanmean  : cm[b,64,64] = mean_c low      (GAM softmax collapses to this)
//   K1 prep_w1   : conv1 weights -> bf16 pairs in exact WMMA A-fragment order
//   K2 stats     : per-channel sum/sumsq of fg/bg (bilinear recompute in LDS)
//   K3 bnp       : fold BN stats + affine into per-channel scale/shift
//   K4 fusion    : relu(bn1(a*lf*s)) - relu(bn2(b*lf*(1-s))) -> bf16,
//                  channels-last into padded [b][258][258][128] buffer
//   K5 conv1     : implicit GEMM via v_wmma_f32_16x16x32_bf16; A/B tiles DMA'd
//                  by TENSOR_LOAD_TO_LDS (double-buffered, s_wait_tensorcnt)
//   K6 conv2     : 128->2 conv, VALU, NCHW f32 output
// ---------------------------------------------------------------------------

typedef __attribute__((ext_vector_type(16))) __bf16        v16bf;
typedef __attribute__((ext_vector_type(8)))  float         v8f;
typedef __attribute__((ext_vector_type(8)))  unsigned int  v8u;
typedef __attribute__((ext_vector_type(4)))  unsigned int  v4u;
typedef __attribute__((ext_vector_type(8)))  int           v8i;
typedef __attribute__((ext_vector_type(4)))  int           v4i;

#define B_    4
#define C_    128
#define HO    256
#define HP    258                    // padded spatial extent
#define NPIX  (HO*HO)                // 65536 per batch
#define NTOT  (B_*NPIX)              // 262144
#define EPS_  1e-5f
#define SCALE 0.24705882352941178f   // (64-1)/(256-1), matches f32(63/255)

__device__ __forceinline__ unsigned int f2bf(float f) {
  unsigned int u = __float_as_uint(f);
  unsigned int r = u + 0x7fffu + ((u >> 16) & 1u);   // round-to-nearest-even
  return r >> 16;
}
__device__ __forceinline__ float bf2f(unsigned int v) {
  return __uint_as_float(v << 16);
}

// ---- Tensor Data Mover: 2-D tile load into LDS (D# per ISA 08 §8.3/8.4) ----
// group0: count=1, lds_addr, global_addr, type=2("image")
// group1: data_size=4B, pad_enable, pad_interval/amount, dims/strides (u32 units)
__device__ __forceinline__ void tdm_load2d(unsigned int lds_off, unsigned long long gaddr,
                                           unsigned int g1w0, unsigned int td0, unsigned int td1,
                                           unsigned int t0, unsigned int t1, unsigned int str0) {
  v4u g0 = { 1u, lds_off, (unsigned int)gaddr,
             (unsigned int)((gaddr >> 32) & 0x01ffffffull) | (2u << 30) };
  v8i g1 = { (int)g1w0,
             (int)(td0 << 16),                       // tensor_dim0[15:0]
             (int)((td0 >> 16) | (td1 << 16)),       // tensor_dim0[31:16] | tensor_dim1[15:0]
             (int)((td1 >> 16) | (t0 << 16)),        // tensor_dim1[31:16] | tile_dim0
             (int)t1,                                // tile_dim1 | tile_dim2(0)<<16
             (int)str0,                              // tensor_dim0_stride[31:0]
             0, 0 };
  v4i gz = { 0, 0, 0, 0 };
#if defined(__clang_major__) && (__clang_major__ >= 23)
  v8i gz8 = { 0, 0, 0, 0, 0, 0, 0, 0 };
  __builtin_amdgcn_tensor_load_to_lds(g0, g1, gz, gz, gz8, 0);
#else
  __builtin_amdgcn_tensor_load_to_lds(g0, g1, gz, gz, 0);
#endif
}
// g1 word0: data_size=2(4B)<<16 | pad_enable<<20 | pad_interval<<22 | pad_amount<<25
#define G1W0_A ((2u<<16) | (1u<<20) | (2u<<22) | (1u<<25))  // every 8 DW pad 2 -> stride 10
#define G1W0_B ((2u<<16) | (1u<<20) | (3u<<22) | (1u<<25))  // every 16 DW pad 2 -> stride 18

// ---------------- K-1: zero the 1-pixel border of padded fusion -------------
__global__ __launch_bounds__(256) void k_zeropad(unsigned int* __restrict__ fusP32) {
  int e = blockIdx.x * 256 + threadIdx.x;   // 1028 blocks * 256 = 4112 border px * 64 u32
  int lane = e & 63, pb = e >> 6;
  int b = pb / 1028, r = pb % 1028;
  int hp, wp;
  if (r < 258)      { hp = 0;       wp = r; }
  else if (r < 516) { hp = 257;     wp = r - 258; }
  else if (r < 772) { hp = r - 515; wp = 0; }
  else              { hp = r - 771; wp = 257; }
  fusP32[((size_t)(b * HP + hp) * HP + wp) * 64 + lane] = 0u;
}

// ---------------- K0: per-batch channel mean of low_feature -----------------
__global__ __launch_bounds__(256) void k_chanmean(const float* __restrict__ low,
                                                  float* __restrict__ cm) {
  int idx = blockIdx.x * 256 + threadIdx.x;      // B*4096 = 16384
  int b = idx >> 12, p = idx & 4095;
  const float* src = low + (size_t)b * C_ * 4096 + p;
  float s = 0.f;
  #pragma unroll 4
  for (int c = 0; c < C_; ++c) s += src[c * 4096];
  cm[idx] = s * (1.0f / C_);
}

// ---------------- K1: pack conv1 weights in A-fragment order ----------------
// Layout: [step s][m][g][r] u32, so each lane's A fragment is 8 consecutive
// dwords in LDS (after TDM pad, row stride 10).  kp_local = (r&3)+4g+8(r>>2)
// per the 16-bit A 16x32 VGPR table; K order is tap-major: kpg = tap*64+cpair.
__global__ __launch_bounds__(256) void k_prep_w1(const float* __restrict__ w1,
                                                 unsigned int* __restrict__ wpk) {
  int id = blockIdx.x * 256 + threadIdx.x;   // 36*128*2*8 = 73728
  int r = id & 7, g = (id >> 3) & 1, m = (id >> 4) & 127, s = id >> 11;
  int kpl = (r & 3) + (g << 2) + ((r >> 2) << 3);
  int kpg = s * 16 + kpl;
  int t = kpg >> 6, cp = kpg & 63;           // tap, channel-pair
  float a  = w1[(m * 128 + 2 * cp    ) * 9 + t];
  float bb = w1[(m * 128 + 2 * cp + 1) * 9 + t];
  wpk[id] = f2bf(a) | (f2bf(bb) << 16);
}

// ---------------- K2: BN batch statistics (one block per channel) -----------
__global__ __launch_bounds__(256) void k_stats(const float* __restrict__ low,
                                               const float* __restrict__ cm,
                                               const float* __restrict__ x,
                                               const float* __restrict__ gamma,
                                               const float* __restrict__ alpha,
                                               const float* __restrict__ beta,
                                               float* __restrict__ stats) {
  __shared__ float sl[4096];
  __shared__ float sc[4096];
  __shared__ float red[256];
  int c = blockIdx.x, tid = threadIdx.x;
  float ga = gamma[0], al = alpha[0], be = beta[0];
  float s1 = 0.f, q1 = 0.f, s2 = 0.f, q2 = 0.f;
  for (int b = 0; b < B_; ++b) {
    __syncthreads();
    const float* lp = low + ((size_t)b * C_ + c) * 4096;
    const float* cp = cm + (size_t)b * 4096;
    for (int q = 0; q < 16; ++q) {
      sl[tid + q * 256] = lp[tid + q * 256];
      sc[tid + q * 256] = cp[tid + q * 256];
    }
    __syncthreads();
    for (int it = 0; it < 256; ++it) {
      int pix = tid + it * 256;
      int hh = pix >> 8, ww = pix & 255;
      float ph = hh * SCALE; int i0 = (int)ph; float fh = ph - i0; int i1 = min(i0 + 1, 63);
      float pw = ww * SCALE; int j0 = (int)pw; float fw = pw - j0; int j1 = min(j0 + 1, 63);
      float ifh = 1.f - fh, ifw = 1.f - fw;
      float up = (sl[i0*64+j0]*ifh + sl[i1*64+j0]*fh) * ifw +
                 (sl[i0*64+j1]*ifh + sl[i1*64+j1]*fh) * fw;
      float cu = (sc[i0*64+j0]*ifh + sc[i1*64+j0]*fh) * ifw +
                 (sc[i0*64+j1]*ifh + sc[i1*64+j1]*fh) * fw;
      float lf = up + ga * cu;
      float xv = x[(size_t)b * NPIX + pix];
      float sg = 1.f / (1.f + expf(-xv));
      float fg = al * lf * sg;
      float bg = be * lf * (1.f - sg);
      s1 += fg; q1 += fg * fg; s2 += bg; q2 += bg * bg;
    }
  }
  float vals[4] = {s1, q1, s2, q2};
  for (int k = 0; k < 4; ++k) {
    __syncthreads();
    red[tid] = vals[k];
    __syncthreads();
    for (int off = 128; off > 0; off >>= 1) {
      if (tid < off) red[tid] += red[tid + off];
      __syncthreads();
    }
    if (tid == 0) stats[k * C_ + c] = red[0];
  }
}

// ---------------- K3: fold stats into scale/shift ---------------------------
__global__ void k_bnp(const float* __restrict__ stats,
                      const float* __restrict__ g1, const float* __restrict__ b1,
                      const float* __restrict__ g2, const float* __restrict__ b2,
                      float* __restrict__ bnp) {
  int c = threadIdx.x;
  if (c >= C_) return;
  const float invN = 1.0f / (float)NTOT;
  float m1 = stats[c] * invN,       v1 = stats[128 + c] * invN - m1 * m1;
  float m2 = stats[256 + c] * invN, v2 = stats[384 + c] * invN - m2 * m2;
  float s1 = g1[c] * rsqrtf(v1 + EPS_), s2 = g2[c] * rsqrtf(v2 + EPS_);
  bnp[c] = s1;        bnp[128 + c] = b1[c] - m1 * s1;
  bnp[256 + c] = s2;  bnp[384 + c] = b2[c] - m2 * s2;
}

// ---------------- K4: fused upsample+GAM+DEM -> padded bf16 channels-last ---
__global__ __launch_bounds__(256) void k_fusion(const float* __restrict__ low,
                                                const float* __restrict__ cm,
                                                const float* __restrict__ x,
                                                const float* __restrict__ gamma,
                                                const float* __restrict__ alpha,
                                                const float* __restrict__ beta,
                                                const float* __restrict__ bnp,
                                                unsigned short* __restrict__ fus) {
  __shared__ __align__(16) float lowb[128 * 68];   // [c][2 rows][34 cols]
  __shared__ float cmb[68];
  __shared__ float cmup[128], segb[128], fwb[128];
  __shared__ int   jlocb[128];
  __shared__ float bnl[512];
  __shared__ __align__(16) unsigned short ostg[128 * 128];  // [w][c]
  int blk = blockIdx.x, tid = threadIdx.x;
  int b = blk >> 9, rem = blk & 511, hh = rem >> 1, w0 = (rem & 1) << 7;
  float ph = hh * SCALE; int i0 = (int)ph; float fh = ph - i0; int i1 = min(i0 + 1, 63);
  int jb = (int)(w0 * SCALE);

  const float* lbase = low + (size_t)b * C_ * 4096;
  for (int q = 0; q < 34; ++q) {               // 34*256 = 8704 = 128*2*34
    int e = tid + q * 256;
    int c = e / 68, r2 = e % 68, row = r2 / 34, jj = r2 % 34;
    int ir = row ? i1 : i0;
    lowb[e] = lbase[(c * 64 + ir) * 64 + min(jb + jj, 63)];
  }
  if (tid < 68) {
    int row = tid / 34, jj = tid % 34, ir = row ? i1 : i0;
    cmb[tid] = cm[(size_t)b * 4096 + ir * 64 + min(jb + jj, 63)];
  }
  if (tid < 128) {
    int w = w0 + tid;
    float pw = w * SCALE; int j0 = (int)pw;
    jlocb[tid] = j0 - jb;
    fwb[tid] = pw - j0;
    float xv = x[(size_t)b * NPIX + hh * 256 + w];
    segb[tid] = 1.f / (1.f + expf(-xv));
  }
  bnl[tid] = bnp[tid];
  bnl[tid + 256] = bnp[tid + 256];
  __syncthreads();
  if (tid < 128) {
    int jl = jlocb[tid]; float fw = fwb[tid];
    cmup[tid] = (cmb[jl]*(1.f-fh) + cmb[34+jl]*fh) * (1.f-fw) +
                (cmb[jl+1]*(1.f-fh) + cmb[34+jl+1]*fh) * fw;
  }
  __syncthreads();

  float ga = gamma[0], al = alpha[0], be = beta[0], ifh = 1.f - fh;
  for (int it = 0; it < 64; ++it) {            // 128 px * 128 ch
    int e = it * 256 + tid;
    int c = e & 127, wl = e >> 7;
    int jl = jlocb[wl]; float fw = fwb[wl];
    const float* lb = &lowb[c * 68];
    float up = (lb[jl]*ifh + lb[34+jl]*fh) * (1.f-fw) +
               (lb[jl+1]*ifh + lb[34+jl+1]*fh) * fw;
    float lf = up + ga * cmup[wl];
    float sg = segb[wl];
    float fg = al * lf * sg;
    float bg = be * lf * (1.f - sg);
    float r1 = fmaxf(fg * bnl[c]       + bnl[128 + c], 0.f);
    float r2 = fmaxf(bg * bnl[256 + c] + bnl[384 + c], 0.f);
    ostg[wl * 128 + c] = (unsigned short)f2bf(r1 - r2);
  }
  __syncthreads();
  const uint4* s4 = (const uint4*)ostg;
  uint4* d4 = (uint4*)(fus + ((size_t)(b * HP + hh + 1) * HP + (w0 + 1)) * 128);
  #pragma unroll
  for (int q = 0; q < 8; ++q) d4[tid + q * 256] = s4[tid + q * 256];  // 32 KB
}

// ---------------- K5: conv1 implicit GEMM, TDM-staged, bf16 WMMA ------------
// Block tile: 128 co x 128 px; 8 waves x (16 co x 128 px).  K = 1152 = 36*32.
// Wave0 issues two TENSOR_LOAD_TO_LDS per step (A weights, B pixel tile) into
// double buffers; s_wait_tensorcnt(2) retires the previous pair (in-order).
__global__ __launch_bounds__(256) void k_conv1(const unsigned short* __restrict__ fusP,
                                               const unsigned int* __restrict__ wpk,
                                               const float* __restrict__ bias1,
                                               unsigned short* __restrict__ y) {
  __shared__ __align__(16) unsigned int AldsT[2][2576];      // 256 rows * 10 (pad) + slack
  __shared__ __align__(16) unsigned int BldsT[2][2336];      // 128 rows * 18 (pad) + slack
  __shared__ __align__(16) unsigned short ostg[128 * 130];   // [px][co], padded
  int tid = threadIdx.x, blk = blockIdx.x;
  int b = blk >> 9, rem = blk & 511, hh = rem >> 1, w0 = (rem & 1) << 7;
  int lane = tid & 31;
  int g = lane >> 4, l16 = lane & 15;
  int mrow = ((tid >> 5) << 4) + l16;          // wave*16 + l16
  int n = tid >> 1, half = tid & 1;

  unsigned long long gW = (unsigned long long)(size_t)wpk;
  unsigned long long gF = (unsigned long long)(size_t)fusP;
  unsigned int ldsA0 = (unsigned int)(size_t)(void*)&AldsT[0][0];
  unsigned int ldsA1 = (unsigned int)(size_t)(void*)&AldsT[1][0];
  unsigned int ldsB0 = (unsigned int)(size_t)(void*)&BldsT[0][0];
  unsigned int ldsB1 = (unsigned int)(size_t)(void*)&BldsT[1][0];
  int wave0 = (__builtin_amdgcn_readfirstlane((int)tid) >> 5) == 0;  // scalar branch

  v8f acc[8] = {};

  // prologue: stage step 0
  if (wave0) {
    int t = 0, cp0 = 0, dh = -1, dw = -1;
    tdm_load2d(ldsA0, gW, G1W0_A, 2048u, 1u, 2048u, 1u, 2048u);
    size_t pix = ((size_t)(b * HP + hh + 1 + dh) * HP + (w0 + 1 + dw));
    tdm_load2d(ldsB0, gF + (pix * 64 + (size_t)(cp0 + t * 0)) * 4ull,
               G1W0_B, 16u, 128u, 16u, 128u, 64u);
  }

  for (int s = 0; s < 36; ++s) {
    if (wave0) {
      if (s + 1 < 36) {
        int sn = s + 1;
        int tn = sn >> 2, cp0n = (sn & 3) << 4;
        int dhn = tn / 3 - 1, dwn = tn % 3 - 1;
        unsigned int la = (sn & 1) ? ldsA1 : ldsA0;
        unsigned int lb = (sn & 1) ? ldsB1 : ldsB0;
        tdm_load2d(la, gW + (unsigned long long)sn * 8192ull,
                   G1W0_A, 2048u, 1u, 2048u, 1u, 2048u);
        size_t pixn = ((size_t)(b * HP + hh + 1 + dhn) * HP + (w0 + 1 + dwn));
        tdm_load2d(lb, gF + (pixn * 64 + (size_t)cp0n) * 4ull,
                   G1W0_B, 16u, 128u, 16u, 128u, 64u);
        __builtin_amdgcn_s_wait_tensorcnt(2);   // oldest pair (step s) complete
      } else {
        __builtin_amdgcn_s_wait_tensorcnt(0);
      }
    }
    __syncthreads();

    const unsigned int* Ab = &AldsT[s & 1][0];
    const unsigned int* Bb = &BldsT[s & 1][0];
    v8u au;
    {
      const uint2* ap = (const uint2*)&Ab[(mrow * 2 + g) * 10];
      uint2 q0 = ap[0], q1 = ap[1], q2 = ap[2], q3 = ap[3];
      au[0]=q0.x; au[1]=q0.y; au[2]=q1.x; au[3]=q1.y;
      au[4]=q2.x; au[5]=q2.y; au[6]=q3.x; au[7]=q3.y;
    }
    v16bf av = __builtin_bit_cast(v16bf, au);
    #pragma unroll
    for (int nt = 0; nt < 8; ++nt) {
      const uint2* bp = (const uint2*)&Bb[(nt * 16 + l16) * 18 + (g << 2)];
      uint2 q0 = bp[0], q1 = bp[1], q2 = bp[4], q3 = bp[5];
      v8u bu;
      bu[0]=q0.x; bu[1]=q0.y; bu[2]=q1.x; bu[3]=q1.y;
      bu[4]=q2.x; bu[5]=q2.y; bu[6]=q3.x; bu[7]=q3.y;
      v16bf bv = __builtin_bit_cast(v16bf, bu);
      acc[nt] = __builtin_amdgcn_wmma_f32_16x16x32_bf16(
          false, av, false, bv, (short)0, acc[nt], false, false);
    }
    __syncthreads();
  }

  // epilogue: bias, bf16, transpose via LDS, contiguous channels-last store
  float bs[8];
  #pragma unroll
  for (int r = 0; r < 8; ++r) bs[r] = bias1[((tid >> 5) << 4) + r + (g << 3)];
  #pragma unroll
  for (int nt = 0; nt < 8; ++nt) {
    int nn = nt * 16 + l16;
    #pragma unroll
    for (int r = 0; r < 8; ++r) {
      int co = ((tid >> 5) << 4) + r + (g << 3);   // D layout: M = r + 8*(lane>=16)
      ostg[nn * 130 + co] = (unsigned short)f2bf(acc[nt][r] + bs[r]);
    }
  }
  __syncthreads();
  unsigned short* dst = y + ((size_t)((b * 256 + hh) * 256) + w0 + n) * 128 + half * 64;
  const unsigned int* srcu = (const unsigned int*)&ostg[n * 130 + half * 64];
  #pragma unroll
  for (int q = 0; q < 8; ++q) {
    uint4 v; v.x = srcu[q*4+0]; v.y = srcu[q*4+1]; v.z = srcu[q*4+2]; v.w = srcu[q*4+3];
    ((uint4*)dst)[q] = v;
  }
}

// ---------------- K6: conv2 (128 -> 2), VALU, NCHW f32 output ---------------
__global__ __launch_bounds__(256) void k_conv2(const unsigned short* __restrict__ y,
                                               const float* __restrict__ w2,
                                               const float* __restrict__ b2,
                                               float* __restrict__ out) {
  __shared__ float wl[2 * 128 * 9];
  int tid = threadIdx.x;
  for (int q = 0; q < 9; ++q) wl[tid + q * 256] = w2[tid + q * 256];
  __syncthreads();
  int p = blockIdx.x * 256 + tid;
  int b = p >> 16, hw = p & 65535, hh = hw >> 8, ww = hw & 255;
  float a0 = b2[0], a1 = b2[1];
  for (int t = 0; t < 9; ++t) {
    int hh2 = hh + t / 3 - 1, ww2 = ww + t % 3 - 1;
    if ((unsigned)hh2 >= 256u || (unsigned)ww2 >= 256u) continue;
    const uint4* yp = (const uint4*)(y + ((size_t)((b * 256 + hh2) * 256 + ww2)) * 128);
    const float* w0p = &wl[t];
    const float* w1p = &wl[128 * 9 + t];
    #pragma unroll 4
    for (int c4 = 0; c4 < 16; ++c4) {
      uint4 u = yp[c4];
      int ci = c4 * 8;
      float f0 = bf2f(u.x & 0xffffu), f1 = bf2f(u.x >> 16);
      float f2 = bf2f(u.y & 0xffffu), f3 = bf2f(u.y >> 16);
      float f4 = bf2f(u.z & 0xffffu), f5 = bf2f(u.z >> 16);
      float f6 = bf2f(u.w & 0xffffu), f7 = bf2f(u.w >> 16);
      a0 += w0p[(ci+0)*9]*f0 + w0p[(ci+1)*9]*f1 + w0p[(ci+2)*9]*f2 + w0p[(ci+3)*9]*f3
          + w0p[(ci+4)*9]*f4 + w0p[(ci+5)*9]*f5 + w0p[(ci+6)*9]*f6 + w0p[(ci+7)*9]*f7;
      a1 += w1p[(ci+0)*9]*f0 + w1p[(ci+1)*9]*f1 + w1p[(ci+2)*9]*f2 + w1p[(ci+3)*9]*f3
          + w1p[(ci+4)*9]*f4 + w1p[(ci+5)*9]*f5 + w1p[(ci+6)*9]*f6 + w1p[(ci+7)*9]*f7;
    }
  }
  out[(size_t)(b * 2 + 0) * NPIX + hh * 256 + ww] = a0;
  out[(size_t)(b * 2 + 1) * NPIX + hh * 256 + ww] = a1;
}

// ---------------------------------------------------------------------------
extern "C" void kernel_launch(void* const* d_in, const int* in_sizes, int n_in,
                              void* d_out, int out_size, void* d_ws, size_t ws_size,
                              hipStream_t stream) {
  (void)in_sizes; (void)n_in; (void)out_size; (void)ws_size;
  const float* low   = (const float*)d_in[0];
  const float* x     = (const float*)d_in[1];
  const float* gamma = (const float*)d_in[2];
  const float* alpha = (const float*)d_in[3];
  const float* beta  = (const float*)d_in[4];
  const float* g1    = (const float*)d_in[5];
  const float* b1    = (const float*)d_in[6];
  const float* g2    = (const float*)d_in[7];
  const float* b2    = (const float*)d_in[8];
  const float* w1    = (const float*)d_in[9];
  const float* cb1   = (const float*)d_in[10];
  const float* w2    = (const float*)d_in[11];
  const float* cb2   = (const float*)d_in[12];

  char* ws = (char*)d_ws;
  float*          cm    = (float*)(ws);                      //    65,536 B
  float*          stats = (float*)(ws + 65536);              //     2,048 B
  float*          bnp   = (float*)(ws + 67584);              //     2,048 B
  unsigned int*   wpk   = (unsigned int*)(ws + 69632);       //   294,912 B
  unsigned short* fusP  = (unsigned short*)(ws + 364544);    // 4*258*258*128*2 = 68,161,536 B
  unsigned short* yb    = (unsigned short*)(ws + 68526080);  // 67,108,864 B
  float*          out   = (float*)d_out;

  k_zeropad <<<1028, 256, 0, stream>>>((unsigned int*)fusP);
  k_chanmean<<<64,   256, 0, stream>>>(low, cm);
  k_prep_w1 <<<288,  256, 0, stream>>>(w1, wpk);
  k_stats   <<<128,  256, 0, stream>>>(low, cm, x, gamma, alpha, beta, stats);
  k_bnp     <<<1,    128, 0, stream>>>(stats, g1, b1, g2, b2, bnp);
  k_fusion  <<<2048, 256, 0, stream>>>(low, cm, x, gamma, alpha, beta, bnp, fusP);
  k_conv1   <<<2048, 256, 0, stream>>>(fusP, wpk, cb1, yb);
  k_conv2   <<<1024, 256, 0, stream>>>(yb, w2, cb2, out);
}
